// GraphFiLM_58153857188396
// MI455X (gfx1250) — compile-verified, matrix operand order checked by reading, not run on hardware
//
#include <hip/hip_runtime.h>
#include <hip/hip_bf16.h>

typedef __attribute__((ext_vector_type(16))) __bf16 v16bf;
typedef __attribute__((ext_vector_type(8)))  __bf16 v8bf;
typedef __attribute__((ext_vector_type(8)))  float  v8f;

#define DIM 64
#define SWZ_ELEMS (2 * 24 * 32 * 16)   // kstep * tile * lane * elem

// ---------------------------------------------------------------------------
// f32 -> bf16 feature conversion
// ---------------------------------------------------------------------------
__global__ void f32_to_bf16_kernel(const float* __restrict__ src,
                                   __bf16* __restrict__ dst, int n) {
  int t = blockIdx.x * blockDim.x + threadIdx.x;
  if (t < n) dst[t] = (__bf16)src[t];
}

// ---------------------------------------------------------------------------
// zero-fill
// ---------------------------------------------------------------------------
__global__ void zero_f32_kernel(float* __restrict__ p, int n) {
  int t = blockIdx.x * blockDim.x + threadIdx.x;
  if (t < n) p[t] = 0.0f;
}

// ---------------------------------------------------------------------------
// Pre-swizzle concatenated weights [64 x 384] (roles: h | beta | gamma | skip
// | beta_s | gamma_s) into the per-lane WMMA B-operand layout for
// v_wmma_f32_16x16x32_bf16.  Linear index = ((kstep*24 + tile)*32 + lane)*16 + i
// K(lane,i) = 32*kstep + 8*(lane/16) + i + (i>=8 ? 8 : 0)
// ---------------------------------------------------------------------------
__global__ void swizzle_weights_kernel(const float* __restrict__ Wlin,
                                       const float* __restrict__ Wfilm,
                                       const float* __restrict__ Wskip,
                                       const float* __restrict__ Wfskip,
                                       __bf16* __restrict__ Bsw) {
  int idx = blockIdx.x * blockDim.x + threadIdx.x;
  if (idx >= SWZ_ELEMS) return;
  int i     = idx & 15;
  int lane  = (idx >> 4) & 31;
  int rest  = idx >> 9;
  int tile  = rest % 24;
  int kstep = rest / 24;
  int g = lane >> 4, n = lane & 15;
  int K = 32 * kstep + 8 * g + i + (i >= 8 ? 8 : 0);
  int role = tile >> 2, jj = tile & 3;
  int col = 16 * jj + n;                // column 0..63 within role
  float v;
  switch (role) {
    case 0:  v = Wlin  [K * 64  + col];       break;  // h
    case 1:  v = Wfilm [K * 128 + col];       break;  // beta
    case 2:  v = Wfilm [K * 128 + 64 + col];  break;  // gamma
    case 3:  v = Wskip [K * 64  + col];       break;  // skip
    case 4:  v = Wfskip[K * 128 + col];       break;  // beta_s
    default: v = Wfskip[K * 128 + 64 + col];  break;  // gamma_s
  }
  Bsw[idx] = (__bf16)v;
}

// ---------------------------------------------------------------------------
// Fused node GEMM: per 16-row block compute all 6 role outputs with WMMA.
// 4 waves / block, wave j owns columns [16j, 16j+16) of each role.
// Epilogue fuses film bias and the self-term relu(gamma_s*skip + beta_s).
// ---------------------------------------------------------------------------
union AVec { v16bf v; v8bf h[2]; };

__global__ __launch_bounds__(128)
void film_gemm_kernel(const __bf16* __restrict__ xbf,
                      const __bf16* __restrict__ Bsw,
                      const float*  __restrict__ b_film,
                      float* __restrict__ H,    float* __restrict__ Beta,
                      float* __restrict__ Gam,  float* __restrict__ SelfB,
                      int N) {
  const int lane = threadIdx.x & 31;
  const int j    = threadIdx.x >> 5;        // wave id = column sub-tile
  const int m    = lane & 15, g = lane >> 4;
  const int r0   = blockIdx.x * 16;

  int arow = r0 + m;
  if (arow >= N) arow = N - 1;              // clamp loads; EXEC stays full
  const __bf16* xr = xbf + (size_t)arow * DIM;

  v8f zero = {};
  v8f acc[6];
#pragma unroll
  for (int r = 0; r < 6; ++r) acc[r] = zero;

#pragma unroll
  for (int ks = 0; ks < 2; ++ks) {
    AVec a;
    a.h[0] = *(const v8bf*)(xr + 32 * ks + 8 * g);        // K = 32ks+8g .. +7
    a.h[1] = *(const v8bf*)(xr + 32 * ks + 16 + 8 * g);   // K = 32ks+16+8g ..
#pragma unroll
    for (int role = 0; role < 6; ++role) {
      int tile = role * 4 + j;
      const v16bf b =
          *(const v16bf*)(Bsw + (((size_t)ks * 24 + tile) * 32 + lane) * 16);
      acc[role] = __builtin_amdgcn_wmma_f32_16x16x32_bf16(
          false, a.v, false, b, (short)0, acc[role], false, false);
    }
  }

  const int col = j * 16 + m;               // 0..63 within each role
  const float bb = b_film[col];
  const float bg = b_film[64 + col];
#pragma unroll
  for (int r = 0; r < 8; ++r) {
    int R = r0 + r + 8 * g;                 // D-matrix row for accumulator r
    if (R < N) {
      size_t o = (size_t)R * DIM + col;
      H[o]    = acc[0][r];
      Beta[o] = acc[1][r] + bb;
      Gam[o]  = acc[2][r] + bg;
      float s = acc[5][r] * acc[3][r] + acc[4][r];
      SelfB[o] = s > 0.0f ? s : 0.0f;
    }
  }
}

// ---------------------------------------------------------------------------
// Edge degree (once; shared by both layers)
// ---------------------------------------------------------------------------
__global__ void edge_deg_kernel(const int* __restrict__ dst,
                                float* __restrict__ Deg, int E) {
  int e = blockIdx.x * blockDim.x + threadIdx.x;
  if (e < E) unsafeAtomicAdd(&Deg[dst[e]], 1.0f);
}

// ---------------------------------------------------------------------------
// Edge FiLM message + scatter-add: msg = relu(gamma[dst]*h[src] + beta[dst])
// 16 threads per edge, float4 per thread, hardware f32 atomics.
// ---------------------------------------------------------------------------
__global__ void edge_msg_kernel(const int* __restrict__ src,
                                const int* __restrict__ dst,
                                const float* __restrict__ H,
                                const float* __restrict__ Gam,
                                const float* __restrict__ Beta,
                                float* __restrict__ Agg, int E) {
  long long t = (long long)blockIdx.x * blockDim.x + threadIdx.x;
  if (t >= (long long)E * 16) return;
  int e = (int)(t >> 4);
  int c = (int)(t & 15);
  int s = src[e], d = dst[e];
  float4 hv = ((const float4*)(H   + (size_t)s * DIM))[c];
  float4 gv = ((const float4*)(Gam + (size_t)d * DIM))[c];
  float4 bv = ((const float4*)(Beta+ (size_t)d * DIM))[c];
  float4 msg;
  msg.x = fmaxf(gv.x * hv.x + bv.x, 0.0f);
  msg.y = fmaxf(gv.y * hv.y + bv.y, 0.0f);
  msg.z = fmaxf(gv.z * hv.z + bv.z, 0.0f);
  msg.w = fmaxf(gv.w * hv.w + bv.w, 0.0f);
  float* ap = Agg + (size_t)d * DIM + 4 * c;
  unsafeAtomicAdd(ap + 0, msg.x);
  unsafeAtomicAdd(ap + 1, msg.y);
  unsafeAtomicAdd(ap + 2, msg.z);
  unsafeAtomicAdd(ap + 3, msg.w);
}

// ---------------------------------------------------------------------------
// Combine: out = leaky_relu(self + agg/max(deg,1), 0.01)
// ---------------------------------------------------------------------------
__global__ void combine_kernel(const float* __restrict__ SelfB,
                               const float* __restrict__ Agg,
                               const float* __restrict__ Deg,
                               float* __restrict__ Out, int n) {
  int t = blockIdx.x * blockDim.x + threadIdx.x;
  if (t >= n) return;
  int i = t >> 6;
  float d = Deg[i];
  d = d > 1.0f ? d : 1.0f;
  float v = SelfB[t] + Agg[t] / d;
  Out[t] = v > 0.0f ? v : 0.01f * v;
}

// ---------------------------------------------------------------------------
// Final head: out[i] = h2[i,:] @ W_out + b_out
// ---------------------------------------------------------------------------
__global__ void head_kernel(const float* __restrict__ X,
                            const float* __restrict__ Wout,
                            const float* __restrict__ bout,
                            float* __restrict__ out, int N) {
  int i = blockIdx.x * blockDim.x + threadIdx.x;
  if (i >= N) return;
  const float4* xp = (const float4*)(X + (size_t)i * DIM);
  const float4* wp = (const float4*)Wout;
  float acc = 0.0f;
#pragma unroll
  for (int k = 0; k < 16; ++k) {
    float4 a = xp[k], b = wp[k];
    acc += a.x * b.x + a.y * b.y + a.z * b.z + a.w * b.w;
  }
  out[i] = acc + bout[0];
}

// ---------------------------------------------------------------------------
// launch
// ---------------------------------------------------------------------------
extern "C" void kernel_launch(void* const* d_in, const int* in_sizes, int n_in,
                              void* d_out, int out_size, void* d_ws, size_t ws_size,
                              hipStream_t stream) {
  const float* x        = (const float*)d_in[0];
  const int*   ei       = (const int*)  d_in[1];
  const float* Wlin1    = (const float*)d_in[2];
  const float* Wfilm1   = (const float*)d_in[3];
  const float* bfilm1   = (const float*)d_in[4];
  const float* Wskip1   = (const float*)d_in[5];
  const float* Wfskip1  = (const float*)d_in[6];
  const float* Wlin2    = (const float*)d_in[7];
  const float* Wfilm2   = (const float*)d_in[8];
  const float* bfilm2   = (const float*)d_in[9];
  const float* Wskip2   = (const float*)d_in[10];
  const float* Wfskip2  = (const float*)d_in[11];
  const float* Wout     = (const float*)d_in[12];
  const float* bout     = (const float*)d_in[13];

  const int N = in_sizes[0] / DIM;
  const int E = in_sizes[1] / 2;
  const int* srcIdx = ei;
  const int* dstIdx = ei + E;

  // workspace carving (base assumed 256B aligned)
  char* ws = (char*)d_ws;
  const size_t nd = (size_t)N * DIM;
  float* H     = (float*)ws; ws += nd * 4;
  float* Beta  = (float*)ws; ws += nd * 4;
  float* Gam   = (float*)ws; ws += nd * 4;
  float* SelfB = (float*)ws; ws += nd * 4;
  float* Agg   = (float*)ws; ws += nd * 4;
  float* X2    = (float*)ws; ws += nd * 4;
  float* Deg   = (float*)ws; ws += (((size_t)N * 4 + 255) / 256) * 256;
  __bf16* xbf  = (__bf16*)ws; ws += nd * 2;
  __bf16* Bsw  = (__bf16*)ws; ws += (size_t)SWZ_ELEMS * 2;

  const int TB = 256;
  const int ndInt = N * DIM;
  dim3 tb(TB);
  int gN   = (N + TB - 1) / TB;
  int gND  = (ndInt + TB - 1) / TB;
  int gE16 = (int)(((long long)E * 16 + TB - 1) / TB);
  int gSwz = (SWZ_ELEMS + TB - 1) / TB;
  int gGemm = (N + 15) / 16;

  // degree (once) + zero agg
  zero_f32_kernel<<<gN, tb, 0, stream>>>(Deg, N);
  zero_f32_kernel<<<gND, tb, 0, stream>>>(Agg, ndInt);
  edge_deg_kernel<<<(E + TB - 1) / TB, tb, 0, stream>>>(dstIdx, Deg, E);

  // ---- layer 1 ----
  f32_to_bf16_kernel<<<gND, tb, 0, stream>>>(x, xbf, ndInt);
  swizzle_weights_kernel<<<gSwz, tb, 0, stream>>>(Wlin1, Wfilm1, Wskip1, Wfskip1, Bsw);
  film_gemm_kernel<<<gGemm, dim3(128), 0, stream>>>(xbf, Bsw, bfilm1, H, Beta, Gam, SelfB, N);
  edge_msg_kernel<<<gE16, tb, 0, stream>>>(srcIdx, dstIdx, H, Gam, Beta, Agg, E);
  combine_kernel<<<gND, tb, 0, stream>>>(SelfB, Agg, Deg, X2, ndInt);

  // ---- layer 2 ----
  f32_to_bf16_kernel<<<gND, tb, 0, stream>>>(X2, xbf, ndInt);
  zero_f32_kernel<<<gND, tb, 0, stream>>>(Agg, ndInt);
  swizzle_weights_kernel<<<gSwz, tb, 0, stream>>>(Wlin2, Wfilm2, Wskip2, Wfskip2, Bsw);
  film_gemm_kernel<<<gGemm, dim3(128), 0, stream>>>(xbf, Bsw, bfilm2, H, Beta, Gam, SelfB, N);
  edge_msg_kernel<<<gE16, tb, 0, stream>>>(srcIdx, dstIdx, H, Gam, Beta, Agg, E);
  combine_kernel<<<gND, tb, 0, stream>>>(SelfB, Agg, Deg, X2, ndInt);

  // ---- head ----
  head_kernel<<<gN, tb, 0, stream>>>(X2, Wout, bout, (float*)d_out, N);
}
